// GPTQMarlinSparseMoELayer_82076825027368
// MI455X (gfx1250) — matrix operand order repaired and checked by reading, not compiled, not run on hardware
//
#include <hip/hip_runtime.h>
#include <hip/hip_bf16.h>

// ---------------- problem constants (match reference) ----------------
#define E_    8
#define T_    4096
#define D_    1024
#define F_    4096
#define TOPK_ 2
#define GROUP_ 128

// ---------------- GEMM tiling ----------------
#define BM 32      // routed rows per block
#define BN 128     // output cols per block
#define BK 32      // K per WMMA step
#define KPAD 40    // LDS row stride for bf16 B tiles (80B, 16B-aligned)

typedef __attribute__((ext_vector_type(16))) __bf16 v16bf;
typedef __attribute__((ext_vector_type(8)))  __bf16 v8bf;
typedef __attribute__((ext_vector_type(8)))  float  v8f;

union AFrag { v16bf v; struct { v8bf lo; v8bf hi; } p; };

// ---------------- CDNA5 async global->LDS path (guarded) ----------------
#if __has_builtin(__builtin_amdgcn_global_load_async_to_lds_b128) && \
    __has_builtin(__builtin_amdgcn_global_load_async_to_lds_b64)  && \
    __has_builtin(__builtin_amdgcn_s_wait_asynccnt)
#define ASYNC_LDS 1
#else
#define ASYNC_LDS 0
#endif

// builtin parameter types (from hipcc diagnostic): vector-size int pointers,
// AS1 for the global source, AS3 for the LDS destination.
typedef int async_v4i __attribute__((vector_size(4 * sizeof(int))));
typedef int async_v2i __attribute__((vector_size(2 * sizeof(int))));
#define AS1_ __attribute__((address_space(1)))
#define AS3_ __attribute__((address_space(3)))

__device__ __forceinline__ void stage_b128(void* lds, const void* g) {
#if ASYNC_LDS
    __builtin_amdgcn_global_load_async_to_lds_b128(
        (AS1_ async_v4i*)g, (AS3_ async_v4i*)lds, 0, 0);
#else
    *(int4*)lds = *(const int4*)g;
#endif
}
__device__ __forceinline__ void stage_b64(void* lds, const void* g) {
#if ASYNC_LDS
    __builtin_amdgcn_global_load_async_to_lds_b64(
        (AS1_ async_v2i*)g, (AS3_ async_v2i*)lds, 0, 0);
#else
    *(long long*)lds = *(const long long*)g;
#endif
}
#if ASYNC_LDS
#define WAIT_ASYNC(n) __builtin_amdgcn_s_wait_asynccnt(n)
#else
#define WAIT_ASYNC(n) ((void)0)
#endif

// pack two floats into one dword of bf16 pair (lowers to v_cvt_pk_bf16_f32)
__device__ __forceinline__ unsigned pk_bf16(float lo, float hi) {
    union { __bf16 h[2]; unsigned u; } p;
    p.h[0] = (__bf16)lo; p.h[1] = (__bf16)hi;
    return p.u;
}

// ============================================================
// 1) gating: softmax -> top2 -> renormalize; histogram counts
// ============================================================
__global__ void moe_zero_counts(int* counts) {
    if (threadIdx.x < E_) counts[threadIdx.x] = 0;
}

__global__ void moe_topk_kernel(const float* __restrict__ gating,
                                int* __restrict__ ids, float* __restrict__ wts,
                                int* __restrict__ counts) {
    int t = blockIdx.x * blockDim.x + threadIdx.x;
    if (t >= T_) return;
    float s[E_];
    float mx = -1e30f;
    #pragma unroll
    for (int e = 0; e < E_; ++e) { s[e] = gating[t * E_ + e]; mx = fmaxf(mx, s[e]); }
    #pragma unroll
    for (int e = 0; e < E_; ++e) s[e] = __expf(s[e] - mx);   // softmax numerator (denom cancels)
    int i0 = 0; float v0 = -1.0f;
    #pragma unroll
    for (int e = 0; e < E_; ++e) { if (s[e] > v0) { v0 = s[e]; i0 = e; } }  // first-on-tie, like top_k
    int i1 = 0; float v1 = -1.0f;
    #pragma unroll
    for (int e = 0; e < E_; ++e) { if (e != i0 && s[e] > v1) { v1 = s[e]; i1 = e; } }
    float rn = 1.0f / (v0 + v1);
    ids[t * TOPK_ + 0] = i0;  wts[t * TOPK_ + 0] = v0 * rn;
    ids[t * TOPK_ + 1] = i1;  wts[t * TOPK_ + 1] = v1 * rn;
    atomicAdd(&counts[i0], 1);
    atomicAdd(&counts[i1], 1);
}

__global__ void moe_scan_offsets(const int* __restrict__ counts, int* __restrict__ offsets) {
    if (threadIdx.x == 0) {
        int a = 0;
        for (int e = 0; e < E_; ++e) { offsets[e] = a; a += counts[e]; }
        offsets[E_] = a;   // == 2T
    }
}

// ============================================================
// 2) deterministic ordered compaction (one wave32 per expert)
// ============================================================
__global__ void moe_route_kernel(const int* __restrict__ ids, const float* __restrict__ wts,
                                 const int* __restrict__ offsets,
                                 int* __restrict__ row_token, float* __restrict__ row_coef,
                                 int* __restrict__ token_rows) {
    const int e = blockIdx.x;
    const int lane = threadIdx.x;     // 32 lanes
    int base = offsets[e];
    for (int t0 = 0; t0 < T_; t0 += 32) {
        int t = t0 + lane;
        #pragma unroll
        for (int j = 0; j < TOPK_; ++j) {
            bool pred = (ids[t * TOPK_ + j] == e);
            unsigned long long m = __ballot(pred);
            int rank = __popcll(m & ((1ull << lane) - 1ull));
            if (pred) {
                int row = base + rank;
                row_token[row] = t;
                row_coef[row]  = wts[t * TOPK_ + j];
                token_rows[t * TOPK_ + j] = row;
            }
            base += __popcll(m);
        }
    }
}

// ============================================================
// 3) GEMM1: act = silu(x @ w1_gate) * (x @ w1_up), routed rows, bf16 WMMA
//    Async double-buffered raw tile staging in LDS; dequant LDS->LDS.
//    grid: (T/BM, F/BN, E), 256 threads (8 waves)
// ============================================================
__global__ __launch_bounds__(256)
void moe_gemm1_kernel(const float* __restrict__ x, const int* __restrict__ w1q,
                      const float* __restrict__ w1s, const int* __restrict__ offsets,
                      const int* __restrict__ row_token, __bf16* __restrict__ act) {
    const int e      = blockIdx.z;
    const int cnt    = offsets[e + 1] - offsets[e];
    const int mBase  = blockIdx.x * BM;
    if (mBase >= cnt) return;
    const int rowOff = offsets[e];
    const int colG   = blockIdx.y * BN;        // gate column base
    const int colU   = F_ + colG;              // up column base (w1 = gate || up)

    // double-buffered raw staging + single-buffered WMMA operand tiles
    __shared__ __align__(16) float  rawA [2][BM][BK];     // 8 KB (x rows, fp32)
    __shared__ __align__(16) int    rawBg[2][BK][BN];     // 32 KB (int4 codes in int32)
    __shared__ __align__(16) int    rawBu[2][BK][BN];     // 32 KB
    __shared__ __align__(16) __bf16 ldsA [BM][BK];        // 2 KB
    __shared__ __align__(16) __bf16 ldsBg[BN][KPAD];      // 10 KB, transposed [n][k]
    __shared__ __align__(16) __bf16 ldsBu[BN][KPAD];      // 10 KB
    __shared__ int srcTok[BM];

    const int tid   = threadIdx.x;
    const int lane  = tid & 31;
    const int wid   = tid >> 5;
    const int waveM = wid >> 2;   // 0..1
    const int waveN = wid & 3;    // 0..3
    const int half  = lane >> 4;
    const int nl    = lane & 15;

    if (tid < BM) {
        int m = (mBase + tid < cnt) ? (mBase + tid) : (cnt - 1);  // clamp: fixed async op count
        srcTok[tid] = row_token[rowOff + m];
    }
    __syncthreads();

    const long long w1base   = (long long)e * D_ * (2 * F_);
    const long long sExpBase = (long long)e * (D_ / GROUP_) * (2 * F_);

    // staging decomposition (per tile, per thread: 1 + 4 + 4 = 9 async ops)
    const int amr = tid >> 3;            // A row 0..31
    const int akq = (tid & 7) * 4;       // A col quad (16B)
    const int kb  = tid & 31;            // B k row
    const int seg = tid >> 5;            // B col segment 0..7

    const int NT = D_ / BK;              // 32 K-tiles
    int buf = 0;

    // ---- prologue: stage tile 0 ----
    {
        const long long xrow = (long long)srcTok[amr] * D_;
        stage_b128(&rawA[0][amr][akq], &x[xrow + akq]);
        const long long rG = w1base + (long long)kb * (2 * F_) + colG;
        const long long rU = w1base + (long long)kb * (2 * F_) + colU;
        #pragma unroll
        for (int i = 0; i < 4; ++i) {
            int n = (seg + i * 8) * 4;
            stage_b128(&rawBg[0][kb][n], &w1q[rG + n]);
            stage_b128(&rawBu[0][kb][n], &w1q[rU + n]);
        }
    }

    v8f zacc = {0.f,0.f,0.f,0.f,0.f,0.f,0.f,0.f};
    v8f accG[2] = {zacc, zacc};
    v8f accU[2] = {zacc, zacc};

    for (int it = 0; it < NT; ++it) {
        const int k0 = it * BK;
        // ---- stage tile it+1 into the other buffer, then wait for tile it ----
        if (it + 1 < NT) {
            const int k1 = k0 + BK;
            const long long xrow = (long long)srcTok[amr] * D_;
            stage_b128(&rawA[buf ^ 1][amr][akq], &x[xrow + k1 + akq]);
            const long long rG = w1base + (long long)(k1 + kb) * (2 * F_) + colG;
            const long long rU = w1base + (long long)(k1 + kb) * (2 * F_) + colU;
            #pragma unroll
            for (int i = 0; i < 4; ++i) {
                int n = (seg + i * 8) * 4;
                stage_b128(&rawBg[buf ^ 1][kb][n], &w1q[rG + n]);
                stage_b128(&rawBu[buf ^ 1][kb][n], &w1q[rU + n]);
            }
            WAIT_ASYNC(9);    // 9 newer ops in flight -> tile `it` fully landed
        } else {
            WAIT_ASYNC(0);
        }
        __syncthreads();

        // ---- dequant phase (LDS -> LDS) ----
        // A: fp32 -> packed bf16
        {
            float4 v = *(const float4*)&rawA[buf][amr][akq];
            unsigned* dst = (unsigned*)&ldsA[amr][akq];
            dst[0] = pk_bf16(v.x, v.y);
            dst[1] = pk_bf16(v.z, v.w);
        }
        // B: int4 codes -> (q-8)*scale, k-pairs packed, transposed
        {
            const int kp   = (tid & 15) * 2;   // even k
            const int sg2  = tid >> 4;         // 0..15
            const long long sRow = sExpBase + (long long)(k0 / GROUP_) * (2 * F_);
            #pragma unroll
            for (int i = 0; i < 2; ++i) {
                const int n0 = (sg2 + i * 16) * 4;
                int4 g0 = *(const int4*)&rawBg[buf][kp][n0];
                int4 g1 = *(const int4*)&rawBg[buf][kp + 1][n0];
                int4 u0 = *(const int4*)&rawBu[buf][kp][n0];
                int4 u1 = *(const int4*)&rawBu[buf][kp + 1][n0];
                float4 sg = *(const float4*)&w1s[sRow + colG + n0];
                float4 su = *(const float4*)&w1s[sRow + colU + n0];
                *(unsigned*)&ldsBg[n0 + 0][kp] = pk_bf16((float)(g0.x - 8) * sg.x, (float)(g1.x - 8) * sg.x);
                *(unsigned*)&ldsBg[n0 + 1][kp] = pk_bf16((float)(g0.y - 8) * sg.y, (float)(g1.y - 8) * sg.y);
                *(unsigned*)&ldsBg[n0 + 2][kp] = pk_bf16((float)(g0.z - 8) * sg.z, (float)(g1.z - 8) * sg.z);
                *(unsigned*)&ldsBg[n0 + 3][kp] = pk_bf16((float)(g0.w - 8) * sg.w, (float)(g1.w - 8) * sg.w);
                *(unsigned*)&ldsBu[n0 + 0][kp] = pk_bf16((float)(u0.x - 8) * su.x, (float)(u1.x - 8) * su.x);
                *(unsigned*)&ldsBu[n0 + 1][kp] = pk_bf16((float)(u0.y - 8) * su.y, (float)(u1.y - 8) * su.y);
                *(unsigned*)&ldsBu[n0 + 2][kp] = pk_bf16((float)(u0.z - 8) * su.z, (float)(u1.z - 8) * su.z);
                *(unsigned*)&ldsBu[n0 + 3][kp] = pk_bf16((float)(u0.w - 8) * su.w, (float)(u1.w - 8) * su.w);
            }
        }
        __syncthreads();

        // ---- WMMA phase ----
        // A layout: lane M = lane%16; elems 0..15 = K(half*16..+15) (contiguous)
        AFrag a;
        const int mrow = waveM * 16 + nl;
        a.p.lo = *(const v8bf*)&ldsA[mrow][half * 16];
        a.p.hi = *(const v8bf*)&ldsA[mrow][half * 16 + 8];
        #pragma unroll
        for (int ns = 0; ns < 2; ++ns) {
            const int ncol = waveN * 32 + ns * 16 + nl;
            // B layout: lane N = lane%16; elems 0..15 = K(half*16..+15) (contiguous)
            v16bf bg = *(const v16bf*)&ldsBg[ncol][half * 16];
            v16bf bu = *(const v16bf*)&ldsBu[ncol][half * 16];
            accG[ns] = __builtin_amdgcn_wmma_f32_16x16x32_bf16(
                false, a.v, false, bg, (short)0, accG[ns], false, false);
            accU[ns] = __builtin_amdgcn_wmma_f32_16x16x32_bf16(
                false, a.v, false, bu, (short)0, accU[ns], false, false);
        }
        buf ^= 1;
    }

    // ---- fused epilogue: silu(gate)*up -> act (bf16) ----
    #pragma unroll
    for (int ns = 0; ns < 2; ++ns) {
        #pragma unroll
        for (int v = 0; v < 8; ++v) {
            int m = waveM * 16 + half * 8 + v;   // C layout: lanes16-31 hold M = 8+v
            if (mBase + m < cnt) {
                float g = accG[ns][v];
                float u = accU[ns][v];
                float av = (g / (1.0f + __expf(-g))) * u;   // silu(g) * u
                int f = colG + waveN * 32 + ns * 16 + nl;
                act[(long long)(rowOff + mBase + m) * F_ + f] = (__bf16)av;
            }
        }
    }
}

// ============================================================
// 4) GEMM2: partial[row] = coef(row) * (act[row] @ dequant(w2[e]))
//    A (bf16) async-copied straight into its WMMA staging buffer.
// ============================================================
__global__ __launch_bounds__(256)
void moe_gemm2_kernel(const __bf16* __restrict__ act, const int* __restrict__ w2q,
                      const float* __restrict__ w2s, const int* __restrict__ offsets,
                      const float* __restrict__ row_coef, float* __restrict__ partial) {
    const int e      = blockIdx.z;
    const int cnt    = offsets[e + 1] - offsets[e];
    const int mBase  = blockIdx.x * BM;
    if (mBase >= cnt) return;
    const int rowOff = offsets[e];
    const int colB   = blockIdx.y * BN;

    __shared__ __align__(16) __bf16 ldsA[2][BM][BK];      // 4 KB (direct async dest)
    __shared__ __align__(16) int    rawB[2][BK][BN];      // 32 KB
    __shared__ __align__(16) __bf16 ldsB[BN][KPAD];       // 10 KB
    __shared__ float sCoef[BM];

    const int tid   = threadIdx.x;
    const int lane  = tid & 31;
    const int wid   = tid >> 5;
    const int waveM = wid >> 2;
    const int waveN = wid & 3;
    const int half  = lane >> 4;
    const int nl    = lane & 15;

    if (tid < BM) sCoef[tid] = (mBase + tid < cnt) ? row_coef[rowOff + mBase + tid] : 0.f;

    const long long w2base   = (long long)e * F_ * D_;
    const long long sExpBase = (long long)e * (F_ / GROUP_) * D_;

    const int amr = tid >> 3;            // A row 0..31
    const int akq = (tid & 7) * 4;       // A col quad (4 bf16 = 8B)
    const int kb  = tid & 31;
    const int seg = tid >> 5;
    // clamp row: fixed async op count per wave, duplicated rows never stored
    const long long arow = (long long)(rowOff + ((mBase + amr < cnt) ? (mBase + amr) : (cnt - 1))) * F_;

    const int NT = F_ / BK;              // 128 K-tiles
    int buf = 0;

    // ---- prologue: stage tile 0 (per thread: 1 + 4 = 5 async ops) ----
    {
        stage_b64(&ldsA[0][amr][akq], &act[arow + akq]);
        const long long rB = w2base + (long long)kb * D_ + colB;
        #pragma unroll
        for (int i = 0; i < 4; ++i) {
            int n = (seg + i * 8) * 4;
            stage_b128(&rawB[0][kb][n], &w2q[rB + n]);
        }
    }

    v8f zacc = {0.f,0.f,0.f,0.f,0.f,0.f,0.f,0.f};
    v8f acc[2] = {zacc, zacc};

    for (int it = 0; it < NT; ++it) {
        const int k0 = it * BK;
        if (it + 1 < NT) {
            const int k1 = k0 + BK;
            stage_b64(&ldsA[buf ^ 1][amr][akq], &act[arow + k1 + akq]);
            const long long rB = w2base + (long long)(k1 + kb) * D_ + colB;
            #pragma unroll
            for (int i = 0; i < 4; ++i) {
                int n = (seg + i * 8) * 4;
                stage_b128(&rawB[buf ^ 1][kb][n], &w2q[rB + n]);
            }
            WAIT_ASYNC(5);
        } else {
            WAIT_ASYNC(0);
        }
        __syncthreads();

        // ---- dequant w2 tile (LDS -> LDS), k-pairs packed ----
        {
            const int kp  = (tid & 15) * 2;
            const int sg2 = tid >> 4;
            const long long sRow = sExpBase + (long long)(k0 / GROUP_) * D_ + colB;
            #pragma unroll
            for (int i = 0; i < 2; ++i) {
                const int n0 = (sg2 + i * 16) * 4;
                int4 q0 = *(const int4*)&rawB[buf][kp][n0];
                int4 q1 = *(const int4*)&rawB[buf][kp + 1][n0];
                float4 s = *(const float4*)&w2s[sRow + n0];
                *(unsigned*)&ldsB[n0 + 0][kp] = pk_bf16((float)(q0.x - 8) * s.x, (float)(q1.x - 8) * s.x);
                *(unsigned*)&ldsB[n0 + 1][kp] = pk_bf16((float)(q0.y - 8) * s.y, (float)(q1.y - 8) * s.y);
                *(unsigned*)&ldsB[n0 + 2][kp] = pk_bf16((float)(q0.z - 8) * s.z, (float)(q1.z - 8) * s.z);
                *(unsigned*)&ldsB[n0 + 3][kp] = pk_bf16((float)(q0.w - 8) * s.w, (float)(q1.w - 8) * s.w);
            }
        }
        __syncthreads();

        AFrag a;
        const int mrow = waveM * 16 + nl;
        a.p.lo = *(const v8bf*)&ldsA[buf][mrow][half * 16];
        a.p.hi = *(const v8bf*)&ldsA[buf][mrow][half * 16 + 8];
        #pragma unroll
        for (int ns = 0; ns < 2; ++ns) {
            const int ncol = waveN * 32 + ns * 16 + nl;
            v16bf b = *(const v16bf*)&ldsB[ncol][half * 16];
            acc[ns] = __builtin_amdgcn_wmma_f32_16x16x32_bf16(
                false, a.v, false, b, (short)0, acc[ns], false, false);
        }
        buf ^= 1;
    }

    #pragma unroll
    for (int ns = 0; ns < 2; ++ns) {
        #pragma unroll
        for (int v = 0; v < 8; ++v) {
            int m = waveM * 16 + half * 8 + v;
            if (mBase + m < cnt) {
                int col = colB + waveN * 32 + ns * 16 + nl;
                partial[(long long)(rowOff + mBase + m) * D_ + col] = sCoef[m] * acc[ns][v];
            }
        }
    }
}

// ============================================================
// 5) combine: out[t] = partial[row(t,0)] + partial[row(t,1)]
// ============================================================
__global__ void moe_combine_kernel(const float* __restrict__ partial,
                                   const int* __restrict__ token_rows,
                                   float* __restrict__ out) {
    int i = blockIdx.x * blockDim.x + threadIdx.x;     // over T*D/4
    if (i >= T_ * (D_ / 4)) return;
    int t  = i / (D_ / 4);
    int d4 = i % (D_ / 4);
    long long r0 = token_rows[t * TOPK_ + 0];
    long long r1 = token_rows[t * TOPK_ + 1];
    float4 a = ((const float4*)partial)[r0 * (D_ / 4) + d4];
    float4 b = ((const float4*)partial)[r1 * (D_ / 4) + d4];
    float4 o;
    o.x = a.x + b.x; o.y = a.y + b.y; o.z = a.z + b.z; o.w = a.w + b.w;
    ((float4*)out)[i] = o;
}

// ============================================================
// launch
// ============================================================
extern "C" void kernel_launch(void* const* d_in, const int* in_sizes, int n_in,
                              void* d_out, int out_size, void* d_ws, size_t ws_size,
                              hipStream_t stream) {
    const float* x      = (const float*)d_in[0];
    const float* gating = (const float*)d_in[1];
    const int*   w1q    = (const int*)d_in[2];
    const int*   w2q    = (const int*)d_in[3];
    const float* w1s    = (const float*)d_in[4];
    const float* w2s    = (const float*)d_in[5];
    float* out = (float*)d_out;

    // ---- workspace carve (~96.3 MB) ----
    char* p = (char*)d_ws;
    size_t off = 0;
    auto take = [&](size_t bytes) { char* r = p + off; off = (off + bytes + 255) & ~(size_t)255; return r; };
    int*    topk_ids   = (int*)   take((size_t)T_ * TOPK_ * sizeof(int));
    float*  topk_wts   = (float*) take((size_t)T_ * TOPK_ * sizeof(float));
    int*    counts     = (int*)   take(64 * sizeof(int));
    int*    offsets    = (int*)   take(64 * sizeof(int));
    int*    row_token  = (int*)   take((size_t)T_ * TOPK_ * sizeof(int));
    float*  row_coef   = (float*) take((size_t)T_ * TOPK_ * sizeof(float));
    int*    token_rows = (int*)   take((size_t)T_ * TOPK_ * sizeof(int));
    __bf16* act        = (__bf16*)take((size_t)T_ * TOPK_ * F_ * sizeof(__bf16)); // 64 MB
    float*  partial    = (float*) take((size_t)T_ * TOPK_ * D_ * sizeof(float));  // 32 MB
    (void)ws_size; (void)in_sizes; (void)n_in; (void)out_size;

    moe_zero_counts<<<1, 64, 0, stream>>>(counts);
    moe_topk_kernel<<<T_ / 128, 128, 0, stream>>>(gating, topk_ids, topk_wts, counts);
    moe_scan_offsets<<<1, 32, 0, stream>>>(counts, offsets);
    moe_route_kernel<<<E_, 32, 0, stream>>>(topk_ids, topk_wts, offsets,
                                            row_token, row_coef, token_rows);
    // m fastest-varying so same-n-stripe tiles are co-resident (L2 catches weight re-reads)
    moe_gemm1_kernel<<<dim3(T_ / BM, F_ / BN, E_), 256, 0, stream>>>(
        x, w1q, w1s, offsets, row_token, act);
    moe_gemm2_kernel<<<dim3(T_ / BM, D_ / BN, E_), 256, 0, stream>>>(
        act, w2q, w2s, offsets, row_coef, partial);
    moe_combine_kernel<<<(T_ * (D_ / 4) + 255) / 256, 256, 0, stream>>>(
        partial, token_rows, out);
}
